// TupleAveragePredictor_20461224198282
// MI455X (gfx1250) — matrix operand order repaired and checked by reading, not compiled, not run on hardware
//
#include <hip/hip_runtime.h>

// ---------------------------------------------------------------------------
// Fused tuple-average + concat + MLP (GEMM via v_wmma_f32_16x16x32_bf16)
// M-tile = 64 rows/block: 4 M-subtiles per wave -> 4 WMMAs per B fragment
// ---------------------------------------------------------------------------

typedef __attribute__((ext_vector_type(16))) __bf16 v16bf;
typedef __attribute__((ext_vector_type(8)))  __bf16 v8bf;
typedef __attribute__((ext_vector_type(8)))  float  v8f;

#define N_ROWS    200000
#define IN_F      256
#define K_DIM     512      // 2 * IN_F
#define H_F       512
#define M_TILE    64       // 200000 / 64 = 3125 blocks, exact
#define XS_STRIDE 520      // 512 + 8 bf16 pad -> conflict-free strided LDS loads

__device__ __forceinline__ unsigned short f2bf(float f) {
    unsigned int u = __float_as_uint(f);
    u += 0x7FFFu + ((u >> 16) & 1u);     // round-to-nearest-even
    return (unsigned short)(u >> 16);
}

__device__ __forceinline__ v16bf cat16(v8bf lo, v8bf hi) {
    return __builtin_shufflevector(lo, hi,
        0, 1, 2, 3, 4, 5, 6, 7, 8, 9, 10, 11, 12, 13, 14, 15);
}

// Pre-kernel: W1 f32 [512,512] -> bf16 bits in workspace (512 KB, L2-resident)
__global__ void cvt_w1_kernel(const float* __restrict__ W1,
                              unsigned short* __restrict__ w1bf) {
    int i = blockIdx.x * blockDim.x + threadIdx.x;
    if (i < H_F * K_DIM) w1bf[i] = f2bf(W1[i]);
}

__global__ __launch_bounds__(256)
void fused_mlp_kernel(const float* __restrict__ gnn_h,
                      const int* __restrict__ tuples,
                      const int* __restrict__ triplets,
                      const unsigned short* __restrict__ w1bf,
                      const float* __restrict__ b1,
                      const float* __restrict__ W2,
                      const float* __restrict__ b2,
                      float* __restrict__ out) {
    __shared__ __align__(16) unsigned short xs[M_TILE * XS_STRIDE];  // 65 KB
    __shared__ float part[M_TILE];

    const int t = threadIdx.x;
    if (t < M_TILE) part[t] = 0.0f;

    // ---- Phase 1: gather 10 rows, average, concat target, bf16 -> LDS ----
    {
        const int r  = t >> 2;                 // tile row 0..63 (4 threads/row)
        const int lc = t & 3;
        const long m = (long)blockIdx.x * M_TILE + r;

        int idx[10];
        #pragma unroll
        for (int j = 0; j < 10; ++j) idx[j] = tuples[m * 11 + 1 + j];
        const int tgt = triplets[m * 3 + 2];

        const int c0 = lc * 64;                // this thread's 64 columns
        #pragma unroll
        for (int chunk = 0; chunk < 4; ++chunk) {
            const int c = c0 + chunk * 16;
            float acc[16];
            #pragma unroll
            for (int i = 0; i < 16; ++i) acc[i] = 0.0f;
            for (int j = 0; j < 10; ++j) {
                const float4* src =
                    (const float4*)(gnn_h + (size_t)idx[j] * IN_F + c);
                #pragma unroll
                for (int q = 0; q < 4; ++q) {
                    float4 v = src[q];
                    acc[q * 4 + 0] += v.x; acc[q * 4 + 1] += v.y;
                    acc[q * 4 + 2] += v.z; acc[q * 4 + 3] += v.w;
                }
            }
            unsigned int* dst = (unsigned int*)(xs + r * XS_STRIDE + c);
            #pragma unroll
            for (int q = 0; q < 8; ++q) {
                unsigned int lo = f2bf(acc[2 * q]     * 0.1f);
                unsigned int hi = f2bf(acc[2 * q + 1] * 0.1f);
                dst[q] = lo | (hi << 16);
            }
            // target features -> columns 256..511
            const float4* tsrc =
                (const float4*)(gnn_h + (size_t)tgt * IN_F + c);
            unsigned int* tdst = (unsigned int*)(xs + r * XS_STRIDE + IN_F + c);
            #pragma unroll
            for (int q = 0; q < 4; ++q) {
                float4 v = tsrc[q];
                tdst[2 * q]     = (unsigned)f2bf(v.x) | ((unsigned)f2bf(v.y) << 16);
                tdst[2 * q + 1] = (unsigned)f2bf(v.z) | ((unsigned)f2bf(v.w) << 16);
            }
        }
    }
    __syncthreads();

    // ---- Phase 2: 64x512 @ 512x512 GEMM with bf16 WMMA ----
    const int lane = t & 31;
    const int wave = t >> 5;                   // 8 waves * 4 ntiles = full N=512
    const int lm   = lane & 15;
    const int h8   = (lane >> 4) << 3;         // unused (0 or 8)
    const int h16  = (lane >> 4) << 4;         // B-frag K phase: 0 or 16

    v8f acc[4][4];                             // [ntile][mtile]
    #pragma unroll
    for (int nt = 0; nt < 4; ++nt)
        #pragma unroll
        for (int mt = 0; mt < 4; ++mt) {
            v8f z = {0.f, 0.f, 0.f, 0.f, 0.f, 0.f, 0.f, 0.f};
            acc[nt][mt] = z;
        }

    const unsigned short* xrow[4];
    #pragma unroll
    for (int mt = 0; mt < 4; ++mt)
        xrow[mt] = xs + (lm + 16 * mt) * XS_STRIDE;

    for (int kk = 0; kk < K_DIM; kk += 32) {
        const int ka = kk + h16;
        // A fragments: K = {ka..ka+15} contiguous, matching B fragment
        v16bf a[4];
        #pragma unroll
        for (int mt = 0; mt < 4; ++mt)
            a[mt] = cat16(*(const v8bf*)(xrow[mt] + ka),
                          *(const v8bf*)(xrow[mt] + ka + 8));

        #pragma unroll
        for (int nt = 0; nt < 4; ++nt) {
            const int n = wave * 64 + nt * 16 + lm;
            // B fragment: 16 contiguous K at kk + h16 (row of W1 = col of W1^T)
            const v8bf* pb =
                (const v8bf*)(w1bf + (size_t)n * K_DIM + kk + h16);
            v16bf bf = cat16(pb[0], pb[1]);
            #pragma unroll
            for (int mt = 0; mt < 4; ++mt)
                acc[nt][mt] = __builtin_amdgcn_wmma_f32_16x16x32_bf16(
                    false, a[mt], false, bf, (short)0, acc[nt][mt], false, false);
        }
    }

    // ---- Phase 3: bias + ReLU + dot(W2) + sigmoid, fused epilogue ----
    float w2v[4], b1v[4];
    #pragma unroll
    for (int nt = 0; nt < 4; ++nt) {
        const int n = wave * 64 + nt * 16 + lm;
        w2v[nt] = W2[n];
        b1v[nt] = b1[n];
    }
    const int rbase = ((lane >> 4) << 3);      // 0 or 8 row offset within subtile
    #pragma unroll
    for (int mt = 0; mt < 4; ++mt) {
        float pr[8];
        #pragma unroll
        for (int i = 0; i < 8; ++i) pr[i] = 0.0f;
        #pragma unroll
        for (int nt = 0; nt < 4; ++nt)
            #pragma unroll
            for (int i = 0; i < 8; ++i) {
                float h = fmaxf(acc[nt][mt][i] + b1v[nt], 0.0f);
                pr[i] += h * w2v[nt];
            }
        #pragma unroll
        for (int i = 0; i < 8; ++i) {
            // butterfly sum over the 16 n-columns held in this half-wave
            float v = pr[i];
            v += __shfl_xor(v, 1);
            v += __shfl_xor(v, 2);
            v += __shfl_xor(v, 4);
            v += __shfl_xor(v, 8);
            if (lm == 0)
                atomicAdd(&part[mt * 16 + i + rbase], v);   // ds_add_f32
        }
    }
    __syncthreads();

    if (t < M_TILE) {
        const float v = part[t] + b2[0];
        out[(size_t)blockIdx.x * M_TILE + t] = 1.0f / (1.0f + __expf(-v));
    }
}

extern "C" void kernel_launch(void* const* d_in, const int* in_sizes, int n_in,
                              void* d_out, int out_size, void* d_ws, size_t ws_size,
                              hipStream_t stream) {
    const float* gnn_h    = (const float*)d_in[0];
    const int*   tuples   = (const int*)d_in[1];
    const int*   triplets = (const int*)d_in[2];
    const float* W1       = (const float*)d_in[3];
    const float* b1       = (const float*)d_in[4];
    const float* W2       = (const float*)d_in[5];
    const float* b2       = (const float*)d_in[6];
    unsigned short* w1bf  = (unsigned short*)d_ws;   // 512 KB bf16 W1
    float* out = (float*)d_out;

    cvt_w1_kernel<<<(H_F * K_DIM + 255) / 256, 256, 0, stream>>>(W1, w1bf);
    fused_mlp_kernel<<<N_ROWS / M_TILE, 256, 0, stream>>>(
        gnn_h, tuples, triplets, w1bf, b1, W2, b2, out);
}
